// CRM_71811853189340
// MI455X (gfx1250) — matrix operand order repaired and checked by reading, not compiled, not run on hardware
//
#include <hip/hip_runtime.h>
#include <math.h>

typedef __attribute__((ext_vector_type(16))) _Float16 v16h;
typedef __attribute__((ext_vector_type(8)))  float    v8f;

#define EPSf 1e-5f

static constexpr int kB  = 2;
static constexpr int kC  = 64;
static constexpr int kH  = 64;
static constexpr int kWd = 64;
static constexpr int kHW = 4096;   // H*W == L
static constexpr int kL  = 4096;
static constexpr int kDI = 128;    // mamba inner dim
static constexpr int kNS = 16;     // D_STATE
static constexpr int kDTR = 4;     // dt rank
static constexpr int kRXP = 36;    // dtr + 2*D_STATE
static constexpr int kMIP = 8;

// =====================================================================
// WMMA GEMM: Y[b][m][n] = sum_k W[m][k] * X[b][k][n] (+ bias[m])
// Compile-time K (64 or 128); N % 16 == 0. Each wave: 32 rows x 16 cols
// (two accumulators share one B fragment). GUARD only for M % 32 != 0.
// =====================================================================
template <bool GUARD>
__device__ __forceinline__ v16h loadA_frag(const float* __restrict__ Wm,
                                           int r, int K, int k0, int hi, int M) {
  const bool ok = !GUARD || (r < M);
  const int  rr = GUARD ? (ok ? r : 0) : r;
  const float* base = Wm + (size_t)rr * K + k0 + hi * 8;
  float4 p0 = *(const float4*)(base);
  float4 p1 = *(const float4*)(base + 4);
  float4 p2 = *(const float4*)(base + 16);
  float4 p3 = *(const float4*)(base + 20);
  const float z = ok ? 1.f : 0.f;   // folds away when !GUARD
  v16h a;
  a[0]  = (_Float16)(p0.x * z); a[1]  = (_Float16)(p0.y * z);
  a[2]  = (_Float16)(p0.z * z); a[3]  = (_Float16)(p0.w * z);
  a[4]  = (_Float16)(p1.x * z); a[5]  = (_Float16)(p1.y * z);
  a[6]  = (_Float16)(p1.z * z); a[7]  = (_Float16)(p1.w * z);
  a[8]  = (_Float16)(p2.x * z); a[9]  = (_Float16)(p2.y * z);
  a[10] = (_Float16)(p2.z * z); a[11] = (_Float16)(p2.w * z);
  a[12] = (_Float16)(p3.x * z); a[13] = (_Float16)(p3.y * z);
  a[14] = (_Float16)(p3.z * z); a[15] = (_Float16)(p3.w * z);
  return a;
}

template <int KT, bool GUARD>
__global__ __launch_bounds__(256) void k_gemm_wmma(
    const float* __restrict__ Wm, const float* __restrict__ X,
    const float* __restrict__ bias, float* __restrict__ Y,
    int M, int N)
{
  const int b     = blockIdx.z;
  const float* Xb = X + (size_t)b * KT * N;
  float*       Yb = Y + (size_t)b * M * N;

  const int wave  = threadIdx.x >> 5;
  const int lane  = threadIdx.x & 31;
  const int nTile = blockIdx.x * 8 + wave;
  if (nTile * 16 >= N) return;                 // wave-uniform
  const int row0 = blockIdx.y * 32;
  const int col  = nTile * 16 + (lane & 15);
  const int hi   = lane >> 4;
  const int rA0  = row0 + (lane & 15);
  const int rA1  = rA0 + 16;

  v8f acc0 = {0.f,0.f,0.f,0.f,0.f,0.f,0.f,0.f};
  v8f acc1 = {0.f,0.f,0.f,0.f,0.f,0.f,0.f,0.f};

#pragma unroll
  for (int k0 = 0; k0 < KT; k0 += 32) {
    v16h a0 = loadA_frag<GUARD>(Wm, rA0, KT, k0, hi, M);
    v16h a1 = loadA_frag<GUARD>(Wm, rA1, KT, k0, hi, M);
    // ---- B fragment (32x16 f16): elems 0-7 K=k0+8*hi+s, elems 8-15 K=k0+16+8*hi+(s-8)
    v16h bf;
    const int kb = k0 + hi * 8;
#pragma unroll
    for (int v = 0; v < 16; ++v) {
      bf[v] = (_Float16)Xb[(size_t)(kb + v + (v >= 8 ? 8 : 0)) * N + col];
    }
    acc0 = __builtin_amdgcn_wmma_f32_16x16x32_f16(false, a0, false, bf,
                                                  (short)0, acc0, false, false);
    acc1 = __builtin_amdgcn_wmma_f32_16x16x32_f16(false, a1, false, bf,
                                                  (short)0, acc1, false, false);
  }

#pragma unroll
  for (int t = 0; t < 2; ++t) {
    v8f acc = t ? acc1 : acc0;
#pragma unroll
    for (int v = 0; v < 8; ++v) {
      const int rr = row0 + t * 16 + v + 8 * hi;
      if (!GUARD || rr < M) {
        float o = acc[v] + (bias ? bias[rr] : 0.f);
        Yb[(size_t)rr * N + col] = o;
      }
    }
  }
}

// =====================================================================
// Depthwise 3x3 (SAME, zero pad), optional 2-source channel concat.
// =====================================================================
__global__ __launch_bounds__(256) void k_dwconv(
    const float* __restrict__ S0, const float* __restrict__ S1,
    const float* __restrict__ dw, const float* __restrict__ db,
    float* __restrict__ Y, int C0, int C1, int H, int Wd, size_t total)
{
  size_t i = (size_t)blockIdx.x * blockDim.x + threadIdx.x;
  if (i >= total) return;
  const int C = C0 + C1;
  int w = (int)(i % Wd);
  int h = (int)((i / Wd) % H);
  int c = (int)((i / ((size_t)Wd * H)) % C);
  int b = (int)(i / ((size_t)Wd * H * C));
  const float* S; int cs, Cs;
  if (c < C0) { S = S0; cs = c;        Cs = C0; }
  else        { S = S1; cs = c - C0;   Cs = C1; }
  const float* p = S + ((size_t)b * Cs + cs) * H * Wd;
  float acc = db[c];
#pragma unroll
  for (int ky = 0; ky < 3; ++ky) {
    int hh = h + ky - 1;
    if (hh < 0 || hh >= H) continue;
#pragma unroll
    for (int kx = 0; kx < 3; ++kx) {
      int ww = w + kx - 1;
      if (ww < 0 || ww >= Wd) continue;
      acc += dw[c * 9 + ky * 3 + kx] * p[(size_t)hh * Wd + ww];
    }
  }
  Y[i] = acc;
}

// ---------------- elementwise helpers ----------------
__device__ __forceinline__ float sigm(float x) { return 1.f / (1.f + expf(-x)); }
__device__ __forceinline__ float hswish(float x) {
  float r = x + 3.f; r = fminf(fmaxf(r, 0.f), 6.f); return x * r * (1.f / 6.f);
}

__global__ void k_gelu(const float* X, float* Y, size_t n) {
  size_t i = (size_t)blockIdx.x * blockDim.x + threadIdx.x;
  if (i < n) { float x = X[i]; Y[i] = 0.5f * x * (1.f + erff(x * 0.70710678f)); }
}
__global__ void k_add(const float* A, const float* B_, float* Y, size_t n) {
  size_t i = (size_t)blockIdx.x * blockDim.x + threadIdx.x;
  if (i < n) Y[i] = A[i] + B_[i];
}
__global__ void k_add_scaled(const float* A, const float* B_, const float* S,
                             float* Y, size_t n) {
  size_t i = (size_t)blockIdx.x * blockDim.x + threadIdx.x;
  if (i < n) Y[i] = A[i] + S[0] * B_[i];
}
__global__ void k_sub(const float* A, const float* B_, float* Y, size_t n) {
  size_t i = (size_t)blockIdx.x * blockDim.x + threadIdx.x;
  if (i < n) Y[i] = A[i] - B_[i];
}
__global__ void k_mix(const float* AHp, const float* AVp, const float* Hs,
                      const float* Vs, float* Y, size_t n) {
  size_t i = (size_t)blockIdx.x * blockDim.x + threadIdx.x;
  if (i < n) Y[i] = sigm(AHp[i]) * Hs[i] + sigm(AVp[i]) * Vs[i];
}
__global__ void k_concat(const float* A, const float* B_, float* Y,
                         int C0, int C1, int HW, size_t n) {
  size_t i = (size_t)blockIdx.x * blockDim.x + threadIdx.x;
  if (i >= n) return;
  int hw = (int)(i % HW);
  int c  = (int)((i / HW) % (C0 + C1));
  int b  = (int)(i / ((size_t)HW * (C0 + C1)));
  Y[i] = (c < C0) ? A[((size_t)b * C0 + c) * HW + hw]
                  : B_[((size_t)b * C1 + (c - C0)) * HW + hw];
}
__global__ void k_bn_act(const float* X, const float* g, const float* bt,
                         const float* rm, const float* rv, float* Y,
                         int C, int HW, int hsw, size_t n) {
  size_t i = (size_t)blockIdx.x * blockDim.x + threadIdx.x;
  if (i >= n) return;
  int c = (int)((i / HW) % C);
  float s = g[c] * rsqrtf(rv[c] + EPSf);
  float v = (X[i] - rm[c]) * s + bt[c];
  Y[i] = hsw ? hswish(v) : v;
}
__global__ void k_scale_chan(const float* X, const float* G, float* Y,
                             int C, int HW, size_t n) {
  size_t i = (size_t)blockIdx.x * blockDim.x + threadIdx.x;
  if (i >= n) return;
  int c = (int)((i / HW) % C);
  int b = (int)(i / ((size_t)HW * C));
  Y[i] = X[i] * G[b * C + c];
}
__global__ void k_scale_axis(const float* X, const float* A, float* Y,
                             int H, int Wd, int perH, size_t n) {
  size_t i = (size_t)blockIdx.x * blockDim.x + threadIdx.x;
  if (i >= n) return;
  int w = (int)(i % Wd);
  int h = (int)((i / Wd) % H);
  size_t bc = i / ((size_t)Wd * H);
  float a = perH ? A[bc * H + h] : A[bc * Wd + w];
  Y[i] = X[i] * a;
}

// ---------------- reductions / norms ----------------
__global__ __launch_bounds__(256) void k_gap(const float* X, float* M, int HW) {
  size_t base = (size_t)blockIdx.x * HW;
  float s = 0.f;
  for (int i = threadIdx.x; i < HW; i += blockDim.x) s += X[base + i];
  __shared__ float sh[256];
  sh[threadIdx.x] = s; __syncthreads();
  for (int k = 128; k >= 1; k >>= 1) {
    if (threadIdx.x < k) sh[threadIdx.x] += sh[threadIdx.x + k];
    __syncthreads();
  }
  if (threadIdx.x == 0) M[blockIdx.x] = sh[0] / HW;
}

__global__ void k_meanW(const float* X, float* Y, int Wd, size_t total) {
  size_t i = (size_t)blockIdx.x * blockDim.x + threadIdx.x;
  if (i >= total) return;
  const float* p = X + i * Wd;
  float s = 0.f;
  for (int w = 0; w < Wd; ++w) s += p[w];
  Y[i] = s / Wd;
}
__global__ void k_meanH(const float* X, float* Y, int H, int Wd, size_t total) {
  size_t i = (size_t)blockIdx.x * blockDim.x + threadIdx.x;
  if (i >= total) return;
  int w = (int)(i % Wd);
  size_t bc = i / Wd;
  const float* p = X + bc * H * Wd + w;
  float s = 0.f;
  for (int h = 0; h < H; ++h) s += p[(size_t)h * Wd];
  Y[i] = s / H;
}

__global__ __launch_bounds__(256) void k_gnorm(const float* X, const float* g,
    const float* bt, float* Y, int GROUPS, int CG, int HW, int relu)
{
  int bg = blockIdx.x;
  int grp = bg % GROUPS;
  size_t base = (size_t)bg * CG * HW;
  int n = CG * HW;
  float s = 0.f, q = 0.f;
  for (int i = threadIdx.x; i < n; i += blockDim.x) {
    float v = X[base + i]; s += v; q += v * v;
  }
  __shared__ float shs[256], shq[256];
  shs[threadIdx.x] = s; shq[threadIdx.x] = q; __syncthreads();
  for (int k = 128; k >= 1; k >>= 1) {
    if (threadIdx.x < k) {
      shs[threadIdx.x] += shs[threadIdx.x + k];
      shq[threadIdx.x] += shq[threadIdx.x + k];
    }
    __syncthreads();
  }
  float mean = shs[0] / n;
  float var  = shq[0] / n - mean * mean;
  float inv  = rsqrtf(var + EPSf);
  int c0 = grp * CG;
  for (int i = threadIdx.x; i < n; i += blockDim.x) {
    int c = c0 + i / HW;
    float v = (X[base + i] - mean) * inv * g[c] + bt[c];
    if (relu) v = fmaxf(v, 0.f);
    Y[base + i] = v;
  }
}

__global__ void k_lnorm(const float* X, const float* g, const float* bt,
                        float* Y, int C, int L, size_t total /*B*L*/) {
  size_t i = (size_t)blockIdx.x * blockDim.x + threadIdx.x;
  if (i >= total) return;
  size_t b = i / L;
  int l = (int)(i % L);
  const float* p = X + b * (size_t)C * L + l;
  float s = 0.f;
  for (int c = 0; c < C; ++c) s += p[(size_t)c * L];
  float m = s / C;
  float v = 0.f;
  for (int c = 0; c < C; ++c) { float d = p[(size_t)c * L] - m; v += d * d; }
  float inv = rsqrtf(v / C + EPSf);
  float* o = Y + b * (size_t)C * L + l;
  for (int c = 0; c < C; ++c)
    o[(size_t)c * L] = (p[(size_t)c * L] - m) * inv * g[c] + bt[c];
}

__global__ __launch_bounds__(64) void k_coord_mlp(
    const float* Yin, const float* w1, const float* b1,
    const float* g, const float* bt, const float* rm, const float* rv,
    const float* w2, const float* b2, float* A, int C, int P)
{
  int bp = blockIdx.x;
  int b = bp / P, p = bp % P;
  __shared__ float t[kMIP];
  int c = threadIdx.x;
  if (c < kMIP) {
    float acc = b1[c];
    for (int j = 0; j < C; ++j)
      acc += w1[c * C + j] * Yin[((size_t)b * C + j) * P + p];
    float inv = rsqrtf(rv[c] + EPSf);
    float v = (acc - rm[c]) * (g[c] * inv) + bt[c];
    t[c] = hswish(v);
  }
  __syncthreads();
  float acc = b2[c];
  for (int j = 0; j < kMIP; ++j) acc += w2[c * kMIP + j] * t[j];
  A[((size_t)b * C + c) * P + p] = sigm(acc);
}

__global__ __launch_bounds__(64) void k_ca_mlp(const float* M, const float* w1,
    const float* w2, float* G, int C, int Hd)
{
  __shared__ float m[64];
  __shared__ float t[16];
  int b = blockIdx.x, c = threadIdx.x;
  m[c] = M[b * C + c]; __syncthreads();
  if (c < Hd) {
    float a = 0.f;
    for (int j = 0; j < C; ++j) a += w1[c * C + j] * m[j];
    t[c] = fmaxf(a, 0.f);
  }
  __syncthreads();
  float a = 0.f;
  for (int j = 0; j < Hd; ++j) a += w2[c * Hd + j] * t[j];
  G[b * C + c] = sigm(a);
}

// ---------------- mamba specific ----------------
__global__ void k_conv1d_silu(const float* XZ, const float* cw, const float* cb,
                              float* XS, int DI, int L, int XZROWS, size_t total) {
  size_t i = (size_t)blockIdx.x * blockDim.x + threadIdx.x;
  if (i >= total) return;
  int l = (int)(i % L);
  int d = (int)((i / L) % DI);
  int b = (int)(i / ((size_t)L * DI));
  const float* r = XZ + ((size_t)b * XZROWS + d) * L;
  float acc = cb[d];
#pragma unroll
  for (int k = 0; k < 4; ++k) {
    int ls = l - 3 + k;
    if (ls >= 0) acc += cw[d * 4 + k] * r[ls];
  }
  XS[i] = acc * sigm(acc);
}

__global__ void k_dt(const float* dbc, const float* dtw, const float* dtb,
                     float* DT, int DI, int L, size_t total) {
  size_t i = (size_t)blockIdx.x * blockDim.x + threadIdx.x;
  if (i >= total) return;
  int l = (int)(i % L);
  int d = (int)((i / L) % DI);
  int b = (int)(i / ((size_t)L * DI));
  const float* p = dbc + (size_t)b * kRXP * L + l;
  float acc = dtb[d];
#pragma unroll
  for (int r = 0; r < kDTR; ++r) acc += dtw[d * kDTR + r] * p[(size_t)r * L];
  DT[i] = (acc > 20.f) ? acc : log1pf(expf(acc));
}

__global__ __launch_bounds__(256) void k_scan(
    const float* DT, const float* XS, const float* XZ, const float* dbc,
    const float* A_log, const float* Dp, float* Y,
    int DI, int L, int XZROWS, int nbd)
{
  int tid = blockIdx.x * blockDim.x + threadIdx.x;
  int n = tid & 15;
  int gd = tid >> 4;
  if (gd >= nbd) return;
  int b = gd / DI, d = gd % DI;
  const float* dtp = DT + ((size_t)b * DI + d) * L;
  const float* xsp = XS + ((size_t)b * DI + d) * L;
  const float* zp  = XZ + ((size_t)b * XZROWS + DI + d) * L;
  const float* Bm  = dbc + ((size_t)b * kRXP + kDTR + n) * L;
  const float* Cm  = dbc + ((size_t)b * kRXP + kDTR + kNS + n) * L;
  float A  = -expf(A_log[(size_t)d * kNS + n]);
  float Dv = Dp[d];
  float* yp = Y + ((size_t)b * DI + d) * L;
  float h = 0.f;
  for (int l = 0; l < L; ++l) {
    float dtv = dtp[l];
    float xv  = xsp[l];
    float a   = expf(dtv * A);
    h = a * h + dtv * xv * Bm[l];
    float contrib = h * Cm[l];
    contrib += __shfl_xor(contrib, 8, 16);
    contrib += __shfl_xor(contrib, 4, 16);
    contrib += __shfl_xor(contrib, 2, 16);
    contrib += __shfl_xor(contrib, 1, 16);
    if (n == 0) {
      float yv = contrib + Dv * xv;
      float zv = zp[l];
      yp[l] = yv * (zv * sigm(zv));
    }
  }
}

// =====================================================================
// Host-side parameter walker: setup_inputs() dict insertion order.
// =====================================================================
struct Wlk { void* const* in; int i; const float* n() { return (const float*)in[i++]; } };
struct CoordP { const float *w1,*b1,*g,*b,*rm,*rv,*w2,*b2; };
static CoordP rdCoord(Wlk& w) {
  CoordP p; p.w1=w.n(); p.b1=w.n(); p.g=w.n(); p.b=w.n(); p.rm=w.n(); p.rv=w.n();
  p.w2=w.n(); p.b2=w.n(); return p;
}
struct DscP { const float *dw,*db,*pw,*pb; };
static DscP rdDsc(Wlk& w) { DscP p; p.dw=w.n(); p.db=w.n(); p.pw=w.n(); p.pb=w.n(); return p; }
struct CaP { const float *w1,*w2; };
static CaP rdCa(Wlk& w) { CaP p; p.w1=w.n(); p.w2=w.n(); return p; }
struct CabP { DscP d1,d2; CaP ca; DscP s2d; CaP s2c; };
static CabP rdCab(Wlk& w) {
  CabP p; p.d1=rdDsc(w); p.d2=rdDsc(w); p.ca=rdCa(w); p.s2d=rdDsc(w); p.s2c=rdCa(w); return p;
}
struct MambaP { const float *in_w,*conv_w,*conv_b,*xproj_w,*dt_w,*dt_b,*A_log,*D,*out_w; };
static MambaP rdMamba(Wlk& w) {
  MambaP p; p.in_w=w.n(); p.conv_w=w.n(); p.conv_b=w.n(); p.xproj_w=w.n();
  p.dt_w=w.n(); p.dt_b=w.n(); p.A_log=w.n(); p.D=w.n(); p.out_w=w.n(); return p;
}
struct MlP { const float *ng,*nb; MambaP m; const float *proj_w,*proj_b,*skip; };
static MlP rdMl(Wlk& w) {
  MlP p; p.ng=w.n(); p.nb=w.n(); p.m=rdMamba(w);
  p.proj_w=w.n(); p.proj_b=w.n(); p.skip=w.n(); return p;
}
struct RmbP { const float *n1g,*n1b,*n2g,*n2b; MlP m1,m2; };
static RmbP rdRmb(Wlk& w) {
  RmbP p; p.n1g=w.n(); p.n1b=w.n(); p.n2g=w.n(); p.n2b=w.n();
  p.m1=rdMl(w); p.m2=rdMl(w); return p;
}
struct CompP { const float *pw,*pb; CabP cab; const float *bg,*bb,*brm,*brv,*wh,*bh,*ww,*bw; };
static CompP rdComp(Wlk& w) {
  CompP p; p.pw=w.n(); p.pb=w.n(); p.cab=rdCab(w);
  p.bg=w.n(); p.bb=w.n(); p.brm=w.n(); p.brv=w.n();
  p.wh=w.n(); p.bh=w.n(); p.ww=w.n(); p.bw=w.n(); return p;
}

// =====================================================================
// Orchestration
// =====================================================================
struct Pool {
  float *t64a,*t64b,*r64a,*r64b;
  float *s128,*s256,*s36,*sdt,*sy;
  float *smallY,*smallA,*smallM,*smallG;
};

static inline dim3 gb(size_t n) { return dim3((unsigned)((n + 255) / 256)); }
static const size_t T64  = (size_t)kB * kC * kHW;
static const size_t T128 = (size_t)kB * kDI * kHW;

static void run_gemm(const float* Wm, const float* X, const float* bias,
                     float* Y, int M, int K, int N, hipStream_t s) {
  dim3 grid((N + 127) / 128, (M + 31) / 32, kB);
  const bool guard = (M % 32) != 0;
  if (K == 64) {
    if (guard) k_gemm_wmma<64,  true ><<<grid, 256, 0, s>>>(Wm, X, bias, Y, M, N);
    else       k_gemm_wmma<64,  false><<<grid, 256, 0, s>>>(Wm, X, bias, Y, M, N);
  } else {  // K == 128 at every call site
    if (guard) k_gemm_wmma<128, true ><<<grid, 256, 0, s>>>(Wm, X, bias, Y, M, N);
    else       k_gemm_wmma<128, false><<<grid, 256, 0, s>>>(Wm, X, bias, Y, M, N);
  }
}

static void run_calayer(const CaP& ca, float* X, Pool& P, hipStream_t s) {
  k_gap<<<kB * kC, 256, 0, s>>>(X, P.smallM, kHW);
  k_ca_mlp<<<kB, kC, 0, s>>>(P.smallM, ca.w1, ca.w2, P.smallG, kC, kC / 4);
  k_scale_chan<<<gb(T64), 256, 0, s>>>(X, P.smallG, X, kC, kHW, T64);
}

static void run_cab(const CabP& p, const float* A, float* out, Pool& P, hipStream_t s) {
  k_dwconv<<<gb(T64), 256, 0, s>>>(A, nullptr, p.d1.dw, p.d1.db, P.t64a,
                                   kC, 0, kH, kWd, T64);
  run_gemm(p.d1.pw, P.t64a, p.d1.pb, P.t64b, kC, kC, kHW, s);
  k_gelu<<<gb(T64), 256, 0, s>>>(P.t64b, P.t64b, T64);
  k_dwconv<<<gb(T64), 256, 0, s>>>(P.t64b, nullptr, p.d2.dw, p.d2.db, P.t64a,
                                   kC, 0, kH, kWd, T64);
  run_gemm(p.d2.pw, P.t64a, p.d2.pb, P.r64a, kC, kC, kHW, s);   // r
  run_calayer(p.ca, P.r64a, P, s);
  k_dwconv<<<gb(T128), 256, 0, s>>>(P.r64a, A, p.s2d.dw, p.s2d.db, P.s128,
                                    kC, kC, kH, kWd, T128);
  run_gemm(p.s2d.pw, P.s128, p.s2d.pb, P.t64b, kC, 2 * kC, kHW, s); // f
  run_calayer(p.s2c, P.t64b, P, s);
  k_add<<<gb(T64), 256, 0, s>>>(P.t64b, P.r64a, out, T64);
}

static void run_mamba_layer(const MlP& p, const float* X, float* Y, Pool& P,
                            hipStream_t s) {
  k_lnorm<<<gb((size_t)kB * kL), 256, 0, s>>>(X, p.ng, p.nb, P.t64a, kC, kL,
                                              (size_t)kB * kL);
  run_gemm(p.m.in_w, P.t64a, nullptr, P.s256, 2 * kDI, kC, kL, s);  // XZ
  k_conv1d_silu<<<gb(T128), 256, 0, s>>>(P.s256, p.m.conv_w, p.m.conv_b,
                                         P.s128, kDI, kL, 2 * kDI, T128);
  run_gemm(p.m.xproj_w, P.s128, nullptr, P.s36, kRXP, kDI, kL, s);  // dbc
  k_dt<<<gb(T128), 256, 0, s>>>(P.s36, p.m.dt_w, p.m.dt_b, P.sdt, kDI, kL, T128);
  k_scan<<<(kB * kDI * kNS + 255) / 256, 256, 0, s>>>(
      P.sdt, P.s128, P.s256, P.s36, p.m.A_log, p.m.D, P.sy,
      kDI, kL, 2 * kDI, kB * kDI);
  run_gemm(p.m.out_w, P.sy, nullptr, P.t64b, kC, kDI, kL, s);
  k_add_scaled<<<gb(T64), 256, 0, s>>>(P.t64b, X, p.skip, P.t64a, T64);
  k_lnorm<<<gb((size_t)kB * kL), 256, 0, s>>>(P.t64a, p.ng, p.nb, P.t64b,
                                              kC, kL, (size_t)kB * kL);
  run_gemm(p.proj_w, P.t64b, p.proj_b, Y, kC, kC, kL, s);
}

static void run_res_mamba(const RmbP& p, const float* X, float* Y, Pool& P,
                          hipStream_t s) {
  k_gnorm<<<kB * 8, 256, 0, s>>>(X, p.n1g, p.n1b, P.r64a, 8, kC / 8, kHW, 1);
  run_mamba_layer(p.m1, P.r64a, P.r64b, P, s);
  k_gnorm<<<kB * 8, 256, 0, s>>>(P.r64b, p.n2g, p.n2b, P.r64a, 8, kC / 8, kHW, 1);
  run_mamba_layer(p.m2, P.r64a, P.r64b, P, s);
  k_add<<<gb(T64), 256, 0, s>>>(P.r64b, X, Y, T64);
}

static void run_coord(const CoordP& p, const float* X, float* Y, int perH,
                      Pool& P, hipStream_t s) {
  size_t nPool = (size_t)kB * kC * (perH ? kH : kWd);
  if (perH) k_meanW<<<gb(nPool), 256, 0, s>>>(X, P.smallY, kWd, nPool);
  else      k_meanH<<<gb(nPool), 256, 0, s>>>(X, P.smallY, kH, kWd, nPool);
  int Ppos = perH ? kH : kWd;
  k_coord_mlp<<<kB * Ppos, kC, 0, s>>>(P.smallY, p.w1, p.b1, p.g, p.b, p.rm,
                                       p.rv, p.w2, p.b2, P.smallA, kC, Ppos);
  k_scale_axis<<<gb(T64), 256, 0, s>>>(X, P.smallA, Y, kH, kWd, perH, T64);
}

extern "C" void kernel_launch(void* const* d_in, const int* in_sizes, int n_in,
                              void* d_out, int out_size, void* d_ws, size_t ws_size,
                              hipStream_t stream) {
  (void)in_sizes; (void)n_in; (void)out_size; (void)ws_size;
  Wlk w{d_in, 0};
  const float* x = w.n();
  CoordP V_att_rain = rdCoord(w), H_att_rain = rdCoord(w);
  CabP   V_cab = rdCab(w),        H_cab = rdCab(w);
  RmbP   V_rb = rdRmb(w),         H_rb = rdRmb(w);
  CoordP V_att = rdCoord(w),      H_att = rdCoord(w);
  CabP   Vf_cab = rdCab(w),       Hf_cab = rdCab(w);
  RmbP   Vf_rb = rdRmb(w),        Hf_rb = rdRmb(w);
  CabP   cab3 = rdCab(w),         cab4 = rdCab(w);
  CompP  comp = rdComp(w);

  float* ws = (float*)d_ws;
  size_t off = 0;
  auto alloc = [&](size_t n) { float* p = ws + off; off += n; return p; };
  float* xV  = alloc(T64);  float* xH  = alloc(T64);
  float* V2H = alloc(T64);  float* H2V = alloc(T64);
  float* Hf  = alloc(T64);  float* Vf  = alloc(T64);
  float* tin = alloc(T64);
  Pool P;
  P.t64a = alloc(T64);  P.t64b = alloc(T64);
  P.r64a = alloc(T64);  P.r64b = alloc(T64);
  P.s128 = alloc(T128); P.s256 = alloc(2 * T128);
  P.s36  = alloc((size_t)kB * kRXP * kL);
  P.sdt  = alloc(T128); P.sy = alloc(T128);
  P.smallY = alloc(8192); P.smallA = alloc(8192);
  P.smallM = alloc(256);  P.smallG = alloc(256);

  hipStream_t s = stream;

  run_coord(V_att_rain, x, P.r64b, 1, P, s);
  run_cab(V_cab, P.r64b, tin, P, s);
  run_res_mamba(V_rb, tin, xV, P, s);

  run_coord(H_att_rain, x, P.r64b, 0, P, s);
  run_cab(H_cab, P.r64b, tin, P, s);
  run_res_mamba(H_rb, tin, xH, P, s);

  run_coord(H_att, xV, P.r64b, 0, P, s);
  run_cab(Hf_cab, P.r64b, tin, P, s);
  run_res_mamba(Hf_rb, tin, V2H, P, s);

  run_coord(V_att, xH, P.r64b, 1, P, s);
  run_cab(Vf_cab, P.r64b, tin, P, s);
  run_res_mamba(Vf_rb, tin, H2V, P, s);

  run_cab(cab3, V2H, tin, P, s);
  k_add<<<gb(T64), 256, 0, s>>>(tin, xH, Hf, T64);
  run_cab(cab4, H2V, tin, P, s);
  k_add<<<gb(T64), 256, 0, s>>>(tin, xV, Vf, T64);

  k_concat<<<gb(T128), 256, 0, s>>>(Vf, Hf, P.s128, kC, kC, kHW, T128);
  run_gemm(comp.pw, P.s128, comp.pb, tin, kC, 2 * kC, kHW, s);
  run_cab(comp.cab, tin, xV, P, s);
  k_bn_act<<<gb(T64), 256, 0, s>>>(xV, comp.bg, comp.bb, comp.brm, comp.brv,
                                   xV, kC, kHW, 1, T64);
  run_gemm(comp.ww, xV, comp.bw, P.t64a, kC, kC, kHW, s);
  run_gemm(comp.wh, xV, comp.bh, P.t64b, kC, kC, kHW, s);
  k_mix<<<gb(T64), 256, 0, s>>>(P.t64a, P.t64b, Hf, Vf, P.r64a, T64);
  k_sub<<<gb(T64), 256, 0, s>>>(x, P.r64a, (float*)d_out, T64);
}
